// CrossAttentionLayer_64493228917276
// MI455X (gfx1250) — matrix operand order repaired and checked by reading, not compiled, not run on hardware
//
#include <hip/hip_runtime.h>

// ---------------------------------------------------------------------------
// CDNA5 (gfx1250, wave32) cross-attention layer, bf16 WMMA everywhere.
// B=8, N=2048, Q=256, D=1024, H=16, HD=64.
// ---------------------------------------------------------------------------

typedef __attribute__((ext_vector_type(16))) __bf16 bf16x16;
typedef __attribute__((ext_vector_type(8)))  __bf16 bf16x8;
typedef __attribute__((ext_vector_type(8)))  float  f32x8;

__device__ __forceinline__ f32x8 wmma_bf16(bf16x16 a, bf16x16 b, f32x8 c) {
  // D = A(16x32) * B(32x16) + C, f32 accumulate
  return __builtin_amdgcn_wmma_f32_16x16x32_bf16(
      /*neg_a=*/false, a, /*neg_b=*/false, b,
      /*c_mod=*/(short)0, c, /*reuse_a=*/false, /*reuse_b=*/false);
}

__device__ __forceinline__ bf16x16 combine16(bf16x8 lo, bf16x8 hi) {
  bf16x16 r;
#pragma unroll
  for (int i = 0; i < 8; ++i) { r[i] = lo[i]; r[i + 8] = hi[i]; }
  return r;
}

// A-fragment (16x32 bf16): lane holds row M=lane&15, K = k0+(half?16:0)+{0..15}
// contiguous (two adjacent 16-byte chunks).
__device__ __forceinline__ bf16x16 load_a_frag(const __bf16* rowptr, int k0, int half) {
  const int kb = k0 + (half ? 16 : 0);
  bf16x8 lo = *(const bf16x8*)(rowptr + kb);
  bf16x8 hi = *(const bf16x8*)(rowptr + kb + 8);
  return combine16(lo, hi);
}

// Same A-fragment but converting f32 activations to bf16 in-register.
__device__ __forceinline__ bf16x16 load_a_frag_f32(const float* rowptr, int k0, int half) {
  const int kb = k0 + (half ? 16 : 0);
  f32x8 lo = *(const f32x8*)(rowptr + kb);
  f32x8 hi = *(const f32x8*)(rowptr + kb + 8);
  bf16x16 r;
#pragma unroll
  for (int i = 0; i < 8; ++i) { r[i] = (__bf16)lo[i]; r[i + 8] = (__bf16)hi[i]; }
  return r;
}

// B-fragment (32x16 bf16): lane holds col N=lane&15, K = (half?16:0)+{0..15}
// contiguous -> one 32-byte load (p must already point at the lane's K base).
__device__ __forceinline__ bf16x16 load_b_frag(const __bf16* p) {
  bf16x8 lo = *(const bf16x8*)(p);
  bf16x8 hi = *(const bf16x8*)(p + 8);
  return combine16(lo, hi);
}

// ---------------------------------------------------------------------------
// f32 -> bf16 conversion (weights)
// ---------------------------------------------------------------------------
__global__ __launch_bounds__(256) void cvt_f32_bf16_kernel(
    const float* __restrict__ in, __bf16* __restrict__ out, int n) {
  int i = blockIdx.x * blockDim.x + threadIdx.x;
  if (i < n) out[i] = (__bf16)in[i];
}

// ---------------------------------------------------------------------------
// QKV projection: out = X @ W^T + bias, scattered into per-head layout.
// One wave computes a 16(M) x 64(Dout) tile; K-loop over Din=1024 in steps 32.
// mode 0: out[b, h, s, hd]   (Q and K)
// mode 1: out[b, h, hd, s]   (V, transposed for the P*V B-operand)
// ---------------------------------------------------------------------------
__global__ __launch_bounds__(128) void qkv_proj_kernel(
    const float*  __restrict__ X,     // [M, 1024] f32
    const __bf16* __restrict__ W,     // [1024, 1024] bf16, row-major [dout, din]
    const float*  __restrict__ bias,  // [1024]
    __bf16*       __restrict__ out,
    int M, int seqlen, int mode) {
  const int D = 1024;
  const int lane = threadIdx.x & 31;
  const int half = lane >> 4;
  const int l16  = lane & 15;
  const int wid  = blockIdx.x * (blockDim.x >> 5) + (threadIdx.x >> 5);
  const int mtile = wid >> 4;          // D/64 = 16 column groups
  const int n0    = (wid & 15) * 64;

  const float* xrow = X + (size_t)(mtile * 16 + l16) * D;

  f32x8 c[4] = {f32x8{}, f32x8{}, f32x8{}, f32x8{}};
  for (int k0 = 0; k0 < D; k0 += 32) {
    bf16x16 a = load_a_frag_f32(xrow, k0, half);
#pragma unroll
    for (int j = 0; j < 4; ++j) {
      const __bf16* wp = W + (size_t)(n0 + j * 16 + l16) * D + k0 + (half ? 16 : 0);
      c[j] = wmma_bf16(a, load_b_frag(wp), c[j]);
    }
  }

#pragma unroll
  for (int j = 0; j < 4; ++j) {
    const int col = n0 + j * 16 + l16;
    const float bv = bias[col];
    const int h = col >> 6, hd = col & 63;
#pragma unroll
    for (int r = 0; r < 8; ++r) {
      const int row = mtile * 16 + r + half * 8;
      const int b = row / seqlen, s = row % seqlen;
      const float val = c[j][r] + bv;
      size_t idx;
      if (mode == 0) idx = (((size_t)(b * 16 + h) * seqlen + s) * 64) + hd;
      else           idx = (((size_t)(b * 16 + h) * 64 + hd) * (size_t)seqlen) + s;
      out[idx] = (__bf16)val;
    }
  }
}

// ---------------------------------------------------------------------------
// Flash attention: one wave per (b, h, 16-query tile), online softmax over N.
// Qh [B,H,Q,HD] bf16, Kh [B,H,N,HD] bf16, Vt [B,H,HD,N] bf16,
// AO [B,Q,D] bf16 (ready as A-operand of the output projection).
// ---------------------------------------------------------------------------
__global__ __launch_bounds__(128) void attn_kernel(
    const __bf16* __restrict__ Qh, const __bf16* __restrict__ Kh,
    const __bf16* __restrict__ Vt, __bf16* __restrict__ AO) {
  const int N = 2048, Qn = 256, HD = 64, Dm = 1024;
  const int LROW = 40;                       // padded LDS row stride (bf16)
  __shared__ __bf16 plds[4][16 * LROW];      // per-wave 16x32 P tile

  const int lane = threadIdx.x & 31;
  const int half = lane >> 4;
  const int l16  = lane & 15;
  const int warp = threadIdx.x >> 5;
  const int wid  = blockIdx.x * 4 + warp;
  const int qt = wid & 15;                   // Q/16 = 16 q-tiles
  const int bh = wid >> 4;                   // 0..127

  const __bf16* qrow = Qh + ((size_t)bh * Qn + qt * 16 + l16) * HD;
  const bf16x16 aq0 = load_a_frag(qrow, 0, half);
  const bf16x16 aq1 = load_a_frag(qrow, 32, half);

  f32x8 O0 = {}, O1 = {}, O2 = {}, O3 = {};
  float m[8], l[8];
#pragma unroll
  for (int r = 0; r < 8; ++r) { m[r] = -1e30f; l[r] = 0.f; }

  __bf16* myl = &plds[warp][0];
  const __bf16* kbase = Kh + (size_t)bh * N * HD;
  const __bf16* vbase = Vt + (size_t)bh * HD * N;

  for (int n0 = 0; n0 < N; n0 += 32) {
    // ---- S = (Q K^T) * scale for two 16-wide n sub-tiles -------------------
    f32x8 s0 = {}, s1 = {};
    {
      const __bf16* kp0 = kbase + (size_t)(n0 + l16) * HD + (half ? 16 : 0);
      s0 = wmma_bf16(aq0, load_b_frag(kp0), s0);
      s0 = wmma_bf16(aq1, load_b_frag(kp0 + 32), s0);
      const __bf16* kp1 = kbase + (size_t)(n0 + 16 + l16) * HD + (half ? 16 : 0);
      s1 = wmma_bf16(aq0, load_b_frag(kp1), s1);
      s1 = wmma_bf16(aq1, load_b_frag(kp1 + 32), s1);
    }
    // ---- online softmax; row q = r + 8*half spans 16 lanes -----------------
    float corr[8];
#pragma unroll
    for (int r = 0; r < 8; ++r) {
      const float a = s0[r] * 0.125f, bb = s1[r] * 0.125f;   // 1/sqrt(64)
      float rm = fmaxf(a, bb);
      rm = fmaxf(rm, __shfl_xor(rm, 1, 32));
      rm = fmaxf(rm, __shfl_xor(rm, 2, 32));
      rm = fmaxf(rm, __shfl_xor(rm, 4, 32));
      rm = fmaxf(rm, __shfl_xor(rm, 8, 32));
      const float mn = fmaxf(m[r], rm);
      corr[r] = __expf(m[r] - mn);
      const float p0 = __expf(a - mn);
      const float p1 = __expf(bb - mn);
      float ps = p0 + p1;
      ps += __shfl_xor(ps, 1, 32);
      ps += __shfl_xor(ps, 2, 32);
      ps += __shfl_xor(ps, 4, 32);
      ps += __shfl_xor(ps, 8, 32);
      l[r] = l[r] * corr[r] + ps;
      m[r] = mn;
      const int ro = (r + half * 8) * LROW;                  // C-layout row
      myl[ro + l16]      = (__bf16)p0;
      myl[ro + 16 + l16] = (__bf16)p1;
    }
    __syncthreads();
    // ---- transpose P through LDS into A-fragment layout --------------------
    bf16x16 pA;
    {
      const __bf16* pr = myl + l16 * LROW + (half ? 16 : 0);
      pA = combine16(*(const bf16x8*)pr, *(const bf16x8*)(pr + 8));
    }
    __syncthreads();
    // ---- rescale accumulators, then O += P * V -----------------------------
#pragma unroll
    for (int r = 0; r < 8; ++r) {
      O0[r] *= corr[r]; O1[r] *= corr[r]; O2[r] *= corr[r]; O3[r] *= corr[r];
    }
    const __bf16* vp = vbase + (size_t)l16 * N + n0 + (half ? 16 : 0);
    O0 = wmma_bf16(pA, load_b_frag(vp + 0 * 16 * N), O0);
    O1 = wmma_bf16(pA, load_b_frag(vp + 1 * 16 * N), O1);
    O2 = wmma_bf16(pA, load_b_frag(vp + 2 * 16 * N), O2);
    O3 = wmma_bf16(pA, load_b_frag(vp + 3 * 16 * N), O3);
  }

  // ---- finalize: divide by softmax denominator and store [B,Q,D] bf16 ------
  const int b = bh >> 4, h = bh & 15;
#pragma unroll
  for (int r = 0; r < 8; ++r) {
    const float inv = 1.f / l[r];
    const size_t row = (size_t)b * Qn + qt * 16 + r + half * 8;
    __bf16* op = AO + row * Dm + h * 64 + l16;
    op[0]  = (__bf16)(O0[r] * inv);
    op[16] = (__bf16)(O1[r] * inv);
    op[32] = (__bf16)(O2[r] * inv);
    op[48] = (__bf16)(O3[r] * inv);
  }
}

// ---------------------------------------------------------------------------
// Output projection: out = AO @ W_out^T + b_out + queries   (f32 result)
// ---------------------------------------------------------------------------
__global__ __launch_bounds__(128) void out_proj_kernel(
    const __bf16* __restrict__ A,      // [2048, 1024] bf16
    const __bf16* __restrict__ W,      // [1024, 1024] bf16
    const float*  __restrict__ bias,   // [1024]
    const float*  __restrict__ resid,  // queries, [2048, 1024] f32
    float*        __restrict__ out) {
  const int D = 1024;
  const int lane = threadIdx.x & 31;
  const int half = lane >> 4;
  const int l16  = lane & 15;
  const int wid  = blockIdx.x * (blockDim.x >> 5) + (threadIdx.x >> 5);
  const int mtile = wid >> 4;
  const int n0    = (wid & 15) * 64;

  const __bf16* arow = A + (size_t)(mtile * 16 + l16) * D;

  f32x8 c[4] = {f32x8{}, f32x8{}, f32x8{}, f32x8{}};
  for (int k0 = 0; k0 < D; k0 += 32) {
    bf16x16 a = load_a_frag(arow, k0, half);
#pragma unroll
    for (int j = 0; j < 4; ++j) {
      const __bf16* wp = W + (size_t)(n0 + j * 16 + l16) * D + k0 + (half ? 16 : 0);
      c[j] = wmma_bf16(a, load_b_frag(wp), c[j]);
    }
  }

#pragma unroll
  for (int j = 0; j < 4; ++j) {
    const int col = n0 + j * 16 + l16;
    const float bv = bias[col];
#pragma unroll
    for (int r = 0; r < 8; ++r) {
      const size_t row = (size_t)mtile * 16 + r + half * 8;
      out[row * D + col] = c[j][r] + bv + resid[row * D + col];
    }
  }
}

// ---------------------------------------------------------------------------
// Host launcher
// ---------------------------------------------------------------------------
extern "C" void kernel_launch(void* const* d_in, const int* in_sizes, int n_in,
                              void* d_out, int out_size, void* d_ws, size_t ws_size,
                              hipStream_t stream) {
  (void)in_sizes; (void)n_in; (void)out_size;
  const float* sources = (const float*)d_in[0];  // [8, 2048, 1024]
  const float* queries = (const float*)d_in[1];  // [8, 256, 1024]
  const float* w_in    = (const float*)d_in[2];  // [3072, 1024]
  const float* b_in    = (const float*)d_in[3];  // [3072]
  const float* w_out   = (const float*)d_in[4];  // [1024, 1024]
  const float* b_out   = (const float*)d_in[5];  // [1024]
  float* out = (float*)d_out;                    // [8, 256, 1024]

  // Workspace layout (bytes), total 80 MiB:
  const size_t OFF_WBIN  = 0;                     // 3072*1024 bf16 = 6 MiB
  const size_t OFF_WBOUT = 6291456;               // 1024*1024 bf16 = 2 MiB
  const size_t OFF_QH    = 8388608;               // [8,16,256,64]  bf16 = 4 MiB
  const size_t OFF_KH    = 12582912;              // [8,16,2048,64] bf16 = 32 MiB
  const size_t OFF_VT    = 46137344;              // [8,16,64,2048] bf16 = 32 MiB
  const size_t OFF_AO    = 79691776;              // [2048,1024]    bf16 = 4 MiB
  const size_t NEEDED    = 83886080;
  if (ws_size < NEEDED) return;

  char* ws = (char*)d_ws;
  __bf16* wbin  = (__bf16*)(ws + OFF_WBIN);
  __bf16* wbout = (__bf16*)(ws + OFF_WBOUT);
  __bf16* Qh    = (__bf16*)(ws + OFF_QH);
  __bf16* Kh    = (__bf16*)(ws + OFF_KH);
  __bf16* Vt    = (__bf16*)(ws + OFF_VT);
  __bf16* AO    = (__bf16*)(ws + OFF_AO);

  // 1) weights -> bf16
  cvt_f32_bf16_kernel<<<3145728 / 256, 256, 0, stream>>>(w_in,  wbin,  3145728);
  cvt_f32_bf16_kernel<<<1048576 / 256, 256, 0, stream>>>(w_out, wbout, 1048576);

  // 2) Q/K/V projections (wave = 16x64 tile; 4 waves per block)
  qkv_proj_kernel<<<512,  128, 0, stream>>>(queries, wbin,              b_in,        Qh, 2048,  256,  0);
  qkv_proj_kernel<<<4096, 128, 0, stream>>>(sources, wbin + 1048576,    b_in + 1024, Kh, 16384, 2048, 0);
  qkv_proj_kernel<<<4096, 128, 0, stream>>>(sources, wbin + 2 * 1048576, b_in + 2048, Vt, 16384, 2048, 1);

  // 3) flash attention: 2048 waves = B*H*(Q/16)
  attn_kernel<<<512, 128, 0, stream>>>(Qh, Kh, Vt, AO);

  // 4) output projection + bias + residual
  out_proj_kernel<<<512, 128, 0, stream>>>(AO, wbout, b_out, queries, out);
}
